// MHA_With_RoPE_63891933495306
// MI455X (gfx1250) — matrix operand order repaired and checked by reading, not compiled, not run on hardware
//
#include <hip/hip_runtime.h>
#include <hip/hip_bf16.h>

// ---------------------------------------------------------------------------
// MHA with RoPE for gfx1250 (MI455X).  B=2, S=2048, D=1024, H=16, d=64.
// GEMMs: 128x128 tile / 8-wave WG, BK=64, double-buffered async-to-LDS,
//        RoPE fused into the Q/K GEMM epilogue (no fp32 round-trip).
// Attention: 4 query tiles per WG, double-buffered async K/V tiles in LDS.
// All matrix math: v_wmma_f32_16x16x32_bf16 (wave32).
// ---------------------------------------------------------------------------

typedef __attribute__((ext_vector_type(16))) __bf16 v16bf;
typedef __attribute__((ext_vector_type(4)))  __bf16 v4bf;
typedef __attribute__((ext_vector_type(8)))  float  v8f;

__device__ __forceinline__ v8f wmma_bf16(v16bf a, v16bf b, v8f c) {
  return __builtin_amdgcn_wmma_f32_16x16x32_bf16(
      false, a, false, b, (short)0, c, false, false);
}

// CDNA5 async copy: global -> LDS, 16 bytes per lane, tracked by ASYNCcnt.
__device__ __forceinline__ void async_b128(void* lds, const void* gaddr) {
  unsigned lo = (unsigned)(size_t)lds;  // LDS aperture: addr[31:0] == LDS offset
  asm volatile("global_load_async_to_lds_b128 %0, %1, off"
               :: "v"(lo), "v"(gaddr) : "memory");
}
// Wait until at most N async ops remain outstanding (loads retire in order,
// so threshold B = "newest batch may still be in flight, older batch landed").
template <int N>
__device__ __forceinline__ void wait_async_le() {
  asm volatile("s_wait_asynccnt %0" :: "i"(N) : "memory");
}

// A-matrix fragment: 16x32 (MxK), bf16, row-major src (row stride ld elements).
__device__ __forceinline__ v16bf load_a_frag(const __bf16* base, int ld) {
  int lane = threadIdx.x & 31;
  int m    = lane & 15;
  int kb   = (lane < 16) ? 0 : 16;
  const __bf16* p = base + (size_t)m * ld + kb;
  v16bf f;
#pragma unroll
  for (int e = 0; e < 8; ++e) { f[e] = p[e]; f[8 + e] = p[8 + e]; }
  return f;
}

// B-matrix fragment 32x16 (KxN): column n of B = row n of row-major W[N][K].
__device__ __forceinline__ v16bf load_b_frag_nk(const __bf16* base, int ld) {
  int lane = threadIdx.x & 31;
  int n    = lane & 15;
  int ks   = (lane < 16) ? 0 : 16;
  const __bf16* p = base + (size_t)n * ld + ks;
  v16bf f;
#pragma unroll
  for (int e = 0; e < 16; ++e) f[e] = p[e];
  return f;
}

// B-matrix fragment from row-major K x N source (e.g. V[key][dim]).
__device__ __forceinline__ v16bf load_b_frag_kn(const __bf16* base, int ld) {
  int lane = threadIdx.x & 31;
  int n    = lane & 15;
  int ks   = (lane < 16) ? 0 : 16;
  v16bf f;
#pragma unroll
  for (int e = 0; e < 16; ++e) f[e] = base[(size_t)(ks + e) * ld + n];
  return f;
}

// ---------------------------------------------------------------------------
// fp32 -> bf16 conversion, 4 elements/thread (b128 in, b64 out)
// ---------------------------------------------------------------------------
__global__ void f32_to_bf16_kernel(const float* __restrict__ in,
                                   __bf16* __restrict__ out, int n) {
  int i = (blockIdx.x * 256 + threadIdx.x) * 4;
  if (i < n) {
    float4 v = *(const float4*)(in + i);
    v4bf o;
    o[0] = (__bf16)v.x; o[1] = (__bf16)v.y; o[2] = (__bf16)v.z; o[3] = (__bf16)v.w;
    *(v4bf*)(out + i) = o;
  }
}

// ---------------------------------------------------------------------------
// Blocked GEMM: C[M,N] = A[M,K] * W[N,K]^T.
// 256 threads (8 waves), 128x128 C tile, BK=64 double-buffered in LDS.
// Wave (wm, wn) computes a 32x64 subtile = 2x4 WMMA accumulators.
// MODE 0: fp32 store.  MODE 1: bf16 store.  MODE 2: bf16 store + fused RoPE
// (each wave's 64-col strip is exactly one head; tile nt pairs with nt+2).
// ---------------------------------------------------------------------------
__device__ __forceinline__ void gemm_stage(const __bf16* __restrict__ A,
                                           const __bf16* __restrict__ W,
                                           __bf16* Abuf, __bf16* Wbuf,
                                           int m0, int n0, int K, int k, int tid) {
  // 128 rows x 64 cols bf16 = 16 KB each => 8 x 16B segs per row
#pragma unroll
  for (int s = tid; s < 1024; s += 256) {
    int r = s >> 3, p = s & 7;
    async_b128(&Abuf[r * 64 + p * 8], A + (size_t)(m0 + r) * K + k + p * 8);
    async_b128(&Wbuf[r * 64 + p * 8], W + (size_t)(n0 + r) * K + k + p * 8);
  }
}

template <int MODE>
__global__ __launch_bounds__(256)
void gemm_xwT_kernel(const __bf16* __restrict__ A, const __bf16* __restrict__ W,
                     void* __restrict__ C, const int* __restrict__ pos,
                     int N, int K) {
  __shared__ __bf16 Ald[2][128 * 64];
  __shared__ __bf16 Wld[2][128 * 64];

  int tid  = threadIdx.x;
  int lane = tid & 31;
  int wave = tid >> 5;
  int wm   = wave & 3;   // row group: 32 rows
  int wn   = wave >> 2;  // col group: 64 cols (one head for D=1024/H=16)
  int m0   = blockIdx.y * 128;
  int n0   = blockIdx.x * 128;

  v8f acc[2][4];
#pragma unroll
  for (int mt = 0; mt < 2; ++mt)
#pragma unroll
    for (int nt = 0; nt < 4; ++nt) acc[mt][nt] = {};

  int nchunks = K >> 6;  // K / 64  (>= 2 here)
  gemm_stage(A, W, Ald[0], Wld[0], m0, n0, K, 0, tid);
  gemm_stage(A, W, Ald[1], Wld[1], m0, n0, K, 64, tid);

  for (int i = 0; i < nchunks; ++i) {
    // 8 async instrs per batch: keep newest batch in flight, older one landed
    if (i + 1 < nchunks) wait_async_le<8>(); else wait_async_le<0>();
    __syncthreads();

    const __bf16* Ab = Ald[i & 1];
    const __bf16* Wb = Wld[i & 1];
#pragma unroll
    for (int kk = 0; kk < 64; kk += 32) {
      v16bf af0 = load_a_frag(&Ab[(wm * 32 + 0) * 64 + kk], 64);
      v16bf af1 = load_a_frag(&Ab[(wm * 32 + 16) * 64 + kk], 64);
#pragma unroll
      for (int nt = 0; nt < 4; ++nt) {
        v16bf bf = load_b_frag_nk(&Wb[(wn * 64 + nt * 16) * 64 + kk], 64);
        acc[0][nt] = wmma_bf16(af0, bf, acc[0][nt]);
        acc[1][nt] = wmma_bf16(af1, bf, acc[1][nt]);
      }
    }
    __syncthreads();  // all waves done reading buf(i) before it is overwritten
    if (i + 2 < nchunks)
      gemm_stage(A, W, Ald[i & 1], Wld[i & 1], m0, n0, K, (i + 2) * 64, tid);
  }

  int col = lane & 15;
  int rb  = (lane < 16) ? 0 : 8;

  if (MODE == 2) {
    // fused RoPE: head-local col c pairs with c+32; freq = pos * 1e4^(-c/32)
    float inv0 = __expf(-9.2103403719761836f * ((float)col * (1.0f / 32.0f)));
    float inv1 = __expf(-9.2103403719761836f * (((float)col + 16.0f) * (1.0f / 32.0f)));
    __bf16* Cb = (__bf16*)C;
#pragma unroll
    for (int mt = 0; mt < 2; ++mt)
#pragma unroll
      for (int e = 0; e < 8; ++e) {
        int r = m0 + wm * 32 + mt * 16 + rb + e;   // flattened (b, s)
        float p = (float)pos[r];
        float sn0, cs0, sn1, cs1;
        __sincosf(p * inv0, &sn0, &cs0);
        __sincosf(p * inv1, &sn1, &cs1);
        size_t base = (size_t)r * N + n0 + wn * 64;
        float q0 = acc[mt][0][e], q1 = acc[mt][1][e];
        float q2 = acc[mt][2][e], q3 = acc[mt][3][e];
        Cb[base + col]      = (__bf16)(q0 * cs0 - q2 * sn0);
        Cb[base + col + 32] = (__bf16)(q2 * cs0 + q0 * sn0);
        Cb[base + col + 16] = (__bf16)(q1 * cs1 - q3 * sn1);
        Cb[base + col + 48] = (__bf16)(q3 * cs1 + q1 * sn1);
      }
  } else {
#pragma unroll
    for (int mt = 0; mt < 2; ++mt)
#pragma unroll
      for (int nt = 0; nt < 4; ++nt)
#pragma unroll
        for (int e = 0; e < 8; ++e) {
          size_t row = (size_t)m0 + wm * 32 + mt * 16 + rb + e;
          size_t cix = row * (size_t)N + n0 + wn * 64 + nt * 16 + col;
          if (MODE == 1) ((__bf16*)C)[cix] = (__bf16)acc[mt][nt][e];
          else           ((float*)C)[cix]  = acc[mt][nt][e];
        }
  }
}

// ---------------------------------------------------------------------------
// Flash attention (causal).  4 waves/WG; wave w owns query tile 4*bx + w of
// head (b,h).  32-key blocks double-buffered via async-to-LDS; per block:
// QK^T (4 WMMA), online softmax (16-lane shuffles), P via LDS, PV (4 WMMA).
// ---------------------------------------------------------------------------
__device__ __forceinline__ void attn_stage(const __bf16* __restrict__ K,
                                           const __bf16* __restrict__ V,
                                           __bf16* Kbuf, __bf16* Vbuf,
                                           size_t kvbase, int tid) {
  // 32 keys x 64 dims bf16 = 4 KB each => 8 x 16B segs per 128B row
#pragma unroll
  for (int s = tid; s < 256; s += 128) {
    int r = s >> 3, p = s & 7;
    async_b128(&Kbuf[r * 64 + p * 8], K + kvbase + (size_t)r * 1024 + p * 8);
    async_b128(&Vbuf[r * 64 + p * 8], V + kvbase + (size_t)r * 1024 + p * 8);
  }
}

__global__ __launch_bounds__(128)
void flash_attn_kernel(const __bf16* __restrict__ Q,
                       const __bf16* __restrict__ K,
                       const __bf16* __restrict__ V,
                       __bf16* __restrict__ O, int S) {
  const int D = 1024;  // H * 64
  __shared__ __bf16 Kld[2][32 * 64];
  __shared__ __bf16 Vld[2][32 * 64];
  __shared__ __bf16 Pld[4][16 * 32];

  int tid  = threadIdx.x;
  int lane = tid & 31;
  int w    = tid >> 5;
  int qt   = blockIdx.x * 4 + w;   // this wave's 16-row query tile
  int bh   = blockIdx.y;
  int b    = bh >> 4;
  int h    = bh & 15;
  int col  = lane & 15;
  int rb   = (lane < 16) ? 0 : 8;

  size_t qbase = ((size_t)b * S + (size_t)qt * 16) * D + (size_t)h * 64;
  v16bf qa0 = load_a_frag(Q + qbase, D);
  v16bf qa1 = load_a_frag(Q + qbase + 32, D);

  v8f o0 = {}, o1 = {}, o2 = {}, o3 = {};
  float m[8], l[8];
#pragma unroll
  for (int e = 0; e < 8; ++e) { m[e] = -1e30f; l[e] = 0.0f; }

  const float scale = 0.125f;                 // 1/sqrt(64)
  int my_nkb  = (qt * 16 + 15) / 32 + 1;      // this wave's causal frontier
  int blk_nkb = blockIdx.x * 2 + 2;           // max over the 4 waves (>= 2)

  size_t hb = ((size_t)b * S) * D + (size_t)h * 64;
  attn_stage(K, V, Kld[0], Vld[0], hb, tid);
  attn_stage(K, V, Kld[1], Vld[1], hb + (size_t)32 * D, tid);

  for (int kb = 0; kb < blk_nkb; ++kb) {
    // 4 async instrs per batch
    if (kb + 1 < blk_nkb) wait_async_le<4>(); else wait_async_le<0>();
    __syncthreads();

    if (kb < my_nkb) {
      const __bf16* Kb = Kld[kb & 1];
      const __bf16* Vb = Vld[kb & 1];
      v16bf kb00 = load_b_frag_nk(&Kb[0 * 64 + 0], 64);
      v16bf kb01 = load_b_frag_nk(&Kb[0 * 64 + 32], 64);
      v16bf kb10 = load_b_frag_nk(&Kb[16 * 64 + 0], 64);
      v16bf kb11 = load_b_frag_nk(&Kb[16 * 64 + 32], 64);
      v8f s0 = {}, s1 = {};
      s0 = wmma_bf16(qa0, kb00, s0);
      s0 = wmma_bf16(qa1, kb01, s0);
      s1 = wmma_bf16(qa0, kb10, s1);
      s1 = wmma_bf16(qa1, kb11, s1);

      float corr[8];
#pragma unroll
      for (int e = 0; e < 8; ++e) {
        int qi  = qt * 16 + rb + e;
        int k0g = kb * 32 + col;
        int k1g = kb * 32 + 16 + col;
        float v0 = s0[e] * scale + ((k0g > qi) ? -1e8f : 0.0f);
        float v1 = s1[e] * scale + ((k1g > qi) ? -1e8f : 0.0f);
        float mx = fmaxf(v0, v1);
#pragma unroll
        for (int off = 8; off >= 1; off >>= 1) mx = fmaxf(mx, __shfl_xor(mx, off, 32));
        float mn = fmaxf(m[e], mx);
        float p0 = __expf(v0 - mn);
        float p1 = __expf(v1 - mn);
        float rs = p0 + p1;
#pragma unroll
        for (int off = 8; off >= 1; off >>= 1) rs += __shfl_xor(rs, off, 32);
        corr[e] = __expf(m[e] - mn);
        l[e] = l[e] * corr[e] + rs;
        m[e] = mn;
        int row = rb + e;
        Pld[w][row * 32 + col]      = (__bf16)p0;
        Pld[w][row * 32 + 16 + col] = (__bf16)p1;
      }
#pragma unroll
      for (int e = 0; e < 8; ++e) {
        o0[e] *= corr[e]; o1[e] *= corr[e]; o2[e] *= corr[e]; o3[e] *= corr[e];
      }
      // same-wave DS ordering covers the P write -> read dependency
      v16bf pa  = load_a_frag(&Pld[w][0], 32);
      v16bf vb0 = load_b_frag_kn(&Vb[0],  64);
      v16bf vb1 = load_b_frag_kn(&Vb[16], 64);
      v16bf vb2 = load_b_frag_kn(&Vb[32], 64);
      v16bf vb3 = load_b_frag_kn(&Vb[48], 64);
      o0 = wmma_bf16(pa, vb0, o0);
      o1 = wmma_bf16(pa, vb1, o1);
      o2 = wmma_bf16(pa, vb2, o2);
      o3 = wmma_bf16(pa, vb3, o3);
    }
    __syncthreads();  // all waves done with buf(kb) before it is overwritten
    if (kb + 2 < blk_nkb)
      attn_stage(K, V, Kld[kb & 1], Vld[kb & 1],
                 hb + (size_t)(kb + 2) * 32 * D, tid);
  }

#pragma unroll
  for (int e = 0; e < 8; ++e) {
    int row = rb + e;
    float inv = 1.0f / l[e];
    size_t ob = ((size_t)b * S + (size_t)qt * 16 + row) * D + (size_t)h * 64 + col;
    O[ob + 0]  = (__bf16)(o0[e] * inv);
    O[ob + 16] = (__bf16)(o1[e] * inv);
    O[ob + 32] = (__bf16)(o2[e] * inv);
    O[ob + 48] = (__bf16)(o3[e] * inv);
  }
}

// ---------------------------------------------------------------------------
// Launch
// ---------------------------------------------------------------------------
extern "C" void kernel_launch(void* const* d_in, const int* in_sizes, int n_in,
                              void* d_out, int out_size, void* d_ws, size_t ws_size,
                              hipStream_t stream) {
  const float* x   = (const float*)d_in[0];
  const int*   pos = (const int*)d_in[1];
  const float* wq  = (const float*)d_in[2];
  const float* wk  = (const float*)d_in[3];
  const float* wv  = (const float*)d_in[4];
  const float* wo  = (const float*)d_in[5];

  const int B = 2, S = 2048, D = 1024;
  const int BS = B * S;  // 4096

  char* ws = (char*)d_ws;
  size_t off = 0;
  auto alloc = [&](size_t bytes) -> void* {
    void* p = ws + off;
    off += (bytes + 255) & ~(size_t)255;
    return p;
  };

  __bf16* Xb  = (__bf16*)alloc((size_t)BS * D * 2);
  __bf16* Wqb = (__bf16*)alloc((size_t)D * D * 2);
  __bf16* Wkb = (__bf16*)alloc((size_t)D * D * 2);
  __bf16* Wvb = (__bf16*)alloc((size_t)D * D * 2);
  __bf16* Wob = (__bf16*)alloc((size_t)D * D * 2);
  __bf16* Qr  = (__bf16*)alloc((size_t)BS * D * 2);  // RoPE'd Q (bf16)
  __bf16* Kr  = (__bf16*)alloc((size_t)BS * D * 2);  // RoPE'd K (bf16)
  __bf16* Vb  = (__bf16*)alloc((size_t)BS * D * 2);
  __bf16* Ab  = (__bf16*)alloc((size_t)BS * D * 2);  // attention output

  int nx = BS * D;
  f32_to_bf16_kernel<<<(nx / 4 + 255) / 256, 256, 0, stream>>>(x, Xb, nx);
  int nw = D * D;
  f32_to_bf16_kernel<<<(nw / 4 + 255) / 256, 256, 0, stream>>>(wq, Wqb, nw);
  f32_to_bf16_kernel<<<(nw / 4 + 255) / 256, 256, 0, stream>>>(wk, Wkb, nw);
  f32_to_bf16_kernel<<<(nw / 4 + 255) / 256, 256, 0, stream>>>(wv, Wvb, nw);
  f32_to_bf16_kernel<<<(nw / 4 + 255) / 256, 256, 0, stream>>>(wo, Wob, nw);

  dim3 gg(D / 128, BS / 128);  // (8, 32)
  gemm_xwT_kernel<2><<<gg, 256, 0, stream>>>(Xb, Wqb, Qr, pos, D, D);  // Q + RoPE
  gemm_xwT_kernel<2><<<gg, 256, 0, stream>>>(Xb, Wkb, Kr, pos, D, D);  // K + RoPE
  gemm_xwT_kernel<1><<<gg, 256, 0, stream>>>(Xb, Wvb, Vb, nullptr, D, D);

  flash_attn_kernel<<<dim3(S / 64, B * 16), 128, 0, stream>>>(Qr, Kr, Vb, Ab, S);

  gemm_xwT_kernel<0><<<gg, 256, 0, stream>>>(Ab, Wob, (float*)d_out, nullptr, D, D);
}
